// CQAttention_2181843386790
// MI455X (gfx1250) — matrix operand order repaired and checked by reading, not compiled, not run on hardware
//
#include <hip/hip_runtime.h>

typedef _Float16 half_t;
typedef __attribute__((ext_vector_type(16))) _Float16 v16h;
typedef __attribute__((ext_vector_type(8)))  _Float16 v8h;
typedef __attribute__((ext_vector_type(8)))  float    v8f;

#define B_  32
#define CL_ 1024
#define QL_ 256
#define D_  768

#define WMMA_F16(af, bf, c) \
  __builtin_amdgcn_wmma_f32_16x16x32_f16(false, (af), false, (bf), (short)0, (c), false, false)

// ---------------- prep kernels ----------------

__global__ void __launch_bounds__(256) k_cw(const float* __restrict__ C,
                                            const float* __restrict__ w3,
                                            half_t* __restrict__ CWh, long n) {
  long i = (long)blockIdx.x * blockDim.x + threadIdx.x;
  if (i < n) {
    int d = (int)(i % D_);
    CWh[i] = (half_t)(C[i] * w3[d]);
  }
}

__global__ void __launch_bounds__(256) k_q(const float* __restrict__ Q,
                                           half_t* __restrict__ Qh,
                                           half_t* __restrict__ QhT, int n) {
  int i = blockIdx.x * blockDim.x + threadIdx.x;
  if (i >= n) return;
  float v = Q[i];
  Qh[i] = (half_t)v;
  int d = i % D_;
  int j = (i / D_) % QL_;
  int b = i / (D_ * QL_);
  QhT[((size_t)b * D_ + d) * QL_ + j] = (half_t)v;
}

// one wave per row: out[row] = dot(X[row,:], w[:d])
__global__ void __launch_bounds__(256) k_rowdot(const float* __restrict__ X,
                                                const float* __restrict__ w,
                                                float* __restrict__ out, int rows, int d) {
  int wv   = (blockIdx.x * blockDim.x + threadIdx.x) >> 5;
  int lane = threadIdx.x & 31;
  if (wv >= rows) return;
  const float* xp = X + (size_t)wv * d;
  float s = 0.f;
  for (int k = lane; k < d; k += 32) s += xp[k] * w[k];
  for (int m = 16; m >= 1; m >>= 1) s += __shfl_xor(s, m, 32);
  if (lane == 0) out[wv] = s;
}

// ---------------- score matmul: Sc = clip(a_i + b_j + (C*w3)·Q^T, ±15) ----------------
// One wave computes a 64x64 output block = 4x4 WMMA tiles (register blocked).
// __launch_bounds__(256) -> 8 waves/WGP -> ~512 VGPR budget, no accumulator spills.

__global__ void __launch_bounds__(256) k_score(const half_t* __restrict__ CWh,
                                               const half_t* __restrict__ Qh,
                                               const float* __restrict__ aC,
                                               const float* __restrict__ bQ,
                                               float* __restrict__ Sc) {
  int w    = (blockIdx.x * blockDim.x + threadIdx.x) >> 5;
  int lane = threadIdx.x & 31;
  int jb = w & 3;           // QL/64 = 4 blocks
  int ib = (w >> 2) & 15;   // CL/64 = 16 blocks
  int b  = w >> 6;
  int i0 = ib * 64, j0 = jb * 64;
  int ln = lane & 15, sel = lane >> 4;

  const half_t* abase = CWh + (size_t)(b * CL_ + i0 + ln) * D_;  // A: row i, K=d
  const half_t* bbase = Qh  + (size_t)(b * QL_ + j0 + ln) * D_;  // B: col j, K=d

  v8f acc[4][4] = {};
  for (int dk = 0; dk < D_; dk += 32) {
    v16h bf[4];
#pragma unroll
    for (int tb = 0; tb < 4; ++tb)
      bf[tb] = *(const v16h*)(bbase + (size_t)(16 * tb) * D_ + dk + 16 * sel);
#pragma unroll
    for (int ta = 0; ta < 4; ++ta) {
      const half_t* ar = abase + (size_t)(16 * ta) * D_;
      v8h alo = *(const v8h*)(ar + dk + 16 * sel);
      v8h ahi = *(const v8h*)(ar + dk + 16 * sel + 8);
      v16h af = __builtin_shufflevector(alo, ahi, 0,1,2,3,4,5,6,7,8,9,10,11,12,13,14,15);
#pragma unroll
      for (int tb = 0; tb < 4; ++tb)
        acc[ta][tb] = WMMA_F16(af, bf[tb], acc[ta][tb]);
    }
  }

#pragma unroll
  for (int ta = 0; ta < 4; ++ta) {
#pragma unroll
    for (int tb = 0; tb < 4; ++tb) {
      int j = j0 + 16 * tb + ln;
      float bj = bQ[b * QL_ + j];
#pragma unroll
      for (int r = 0; r < 8; ++r) {
        int i = i0 + 16 * ta + r + 8 * sel;
        float v = acc[ta][tb][r] + aC[b * CL_ + i] + bj;
        v = fminf(fmaxf(v, -15.f), 15.f);
        Sc[(size_t)(b * CL_ + i) * QL_ + j] = v;
      }
    }
  }
}

// ---------------- row softmax (axis=2, mask q_mask) -> S1h (f16) ----------------

__global__ void __launch_bounds__(256) k_srow(const float* __restrict__ Sc,
                                              const int* __restrict__ qmask,
                                              half_t* __restrict__ S1h) {
  int w    = (blockIdx.x * blockDim.x + threadIdx.x) >> 5;  // row over B*CL
  int lane = threadIdx.x & 31;
  int b = w >> 10;  // CL = 1024
  const float* row = Sc + (size_t)w * QL_;
  const int* qm = qmask + b * QL_;

  float x[8], qv[8];
  float m = -1e30f;
  for (int t = 0; t < 8; ++t) {
    int j = lane + 32 * t;
    qv[t] = (float)qm[j];
    x[t]  = row[j] * qv[t];
    m = fmaxf(m, x[t]);
  }
  for (int s = 16; s >= 1; s >>= 1) m = fmaxf(m, __shfl_xor(m, s, 32));
  float e[8], sum = 0.f;
  for (int t = 0; t < 8; ++t) { e[t] = __expf(x[t] - m) * qv[t]; sum += e[t]; }
  for (int s = 16; s >= 1; s >>= 1) sum += __shfl_xor(sum, s, 32);
  float inv = 1.f / (sum + 1e-6f);
  half_t* orow = S1h + (size_t)w * QL_;
  for (int t = 0; t < 8; ++t) orow[lane + 32 * t] = (half_t)(e[t] * inv);
}

// ---------------- column softmax max (axis=1, mask c_mask), split over i-chunks ----------

#define NCHUNK_ 8
#define CHUNKI_ (CL_ / NCHUNK_)   // 128

__global__ void __launch_bounds__(256) k_scolA(const float* __restrict__ Sc,
                                               const int* __restrict__ cmask,
                                               float* __restrict__ pM,
                                               float* __restrict__ pMv,
                                               float* __restrict__ pS) {
  int b = blockIdx.x;
  int c = blockIdx.y;
  int j = threadIdx.x;  // 256 columns
  int ibeg = c * CHUNKI_;
  const float* base = Sc + (size_t)(b * CL_ + ibeg) * QL_ + j;
  const int* cm = cmask + b * CL_ + ibeg;

  float m = -1e30f, mv = -1e30f;
  for (int i = 0; i < CHUNKI_; ++i) {
    float cmv = (float)cm[i];
    float x = base[(size_t)i * QL_] * cmv;   // masked rows give x = 0 (included in max)
    m = fmaxf(m, x);
    mv = (cmv > 0.f) ? fmaxf(mv, x) : mv;
  }
  float s = 0.f;
  for (int i = 0; i < CHUNKI_; ++i) {
    float cmv = (float)cm[i];
    float x = base[(size_t)i * QL_] * cmv;
    s += (cmv > 0.f) ? __expf(x - m) : 0.f;   // independent exps, no serial chain
  }
  int idx = (b * NCHUNK_ + c) * QL_ + j;
  pM[idx] = m; pMv[idx] = mv; pS[idx] = s;
}

__global__ void __launch_bounds__(256) k_scolB(const float* __restrict__ pM,
                                               const float* __restrict__ pMv,
                                               const float* __restrict__ pS,
                                               float* __restrict__ S2max) {
  int b = blockIdx.x;
  int j = threadIdx.x;
  float M = -1e30f, Mv = -1e30f;
  for (int c = 0; c < NCHUNK_; ++c) {
    int idx = (b * NCHUNK_ + c) * QL_ + j;
    M  = fmaxf(M, pM[idx]);
    Mv = fmaxf(Mv, pMv[idx]);
  }
  float S = 0.f;
  for (int c = 0; c < NCHUNK_; ++c) {
    int idx = (b * NCHUNK_ + c) * QL_ + j;
    S += pS[idx] * __expf(pM[idx] - M);
  }
  float emax = (Mv > -1e29f) ? __expf(Mv - M) : 0.f;
  S2max[b * QL_ + j] = emax / (S + 1e-6f);
}

// ---------------- Bvec[b,d] = sum_j S2max[b,j] * Q[b,j,d] ----------------

__global__ void __launch_bounds__(256) k_bvec(const float* __restrict__ Q,
                                              const float* __restrict__ S2max,
                                              float* __restrict__ Bvec) {
  __shared__ float s2[QL_];
  int b = blockIdx.x;
  int d = blockIdx.y * 256 + threadIdx.x;
  s2[threadIdx.x] = S2max[b * QL_ + threadIdx.x];
  __syncthreads();
  float acc = 0.f;
  const float* qb = Q + (size_t)b * QL_ * D_ + d;
  for (int j = 0; j < QL_; ++j) acc += s2[j] * qb[(size_t)j * D_];
  Bvec[b * D_ + d] = acc;
}

// ---------------- A = S1·Q (WMMA, 64x64 register-blocked) + fused output ----------------
// out[b,i,:] = [C, A, C*A, C*Bvec]

__global__ void __launch_bounds__(256) k_out(const half_t* __restrict__ S1h,
                                             const half_t* __restrict__ QhT,
                                             const float* __restrict__ C,
                                             const float* __restrict__ Bvec,
                                             float* __restrict__ out) {
  int w    = (blockIdx.x * blockDim.x + threadIdx.x) >> 5;
  int lane = threadIdx.x & 31;
  int db = w % 12;           // D/64 = 12 blocks
  int ib = (w / 12) & 15;    // CL/64 = 16 blocks
  int b  = w / (12 * 16);
  int i0 = ib * 64, d0 = db * 64;
  int ln = lane & 15, sel = lane >> 4;

  const half_t* abase = S1h + (size_t)(b * CL_ + i0 + ln) * QL_;  // A: row i, K=j
  const half_t* bbase = QhT + (size_t)(b * D_ + d0 + ln) * QL_;   // B: col d, K=j

  v8f acc[4][4] = {};
  for (int jk = 0; jk < QL_; jk += 32) {
    v16h bf[4];
#pragma unroll
    for (int td = 0; td < 4; ++td)
      bf[td] = *(const v16h*)(bbase + (size_t)(16 * td) * QL_ + jk + 16 * sel);
#pragma unroll
    for (int ta = 0; ta < 4; ++ta) {
      const half_t* ar = abase + (size_t)(16 * ta) * QL_;
      v8h alo = *(const v8h*)(ar + jk + 16 * sel);
      v8h ahi = *(const v8h*)(ar + jk + 16 * sel + 8);
      v16h af = __builtin_shufflevector(alo, ahi, 0,1,2,3,4,5,6,7,8,9,10,11,12,13,14,15);
#pragma unroll
      for (int td = 0; td < 4; ++td)
        acc[ta][td] = WMMA_F16(af, bf[td], acc[ta][td]);
    }
  }

#pragma unroll
  for (int td = 0; td < 4; ++td) {
    int d = d0 + 16 * td + ln;
    float bv = Bvec[b * D_ + d];
#pragma unroll
    for (int ta = 0; ta < 4; ++ta) {
#pragma unroll
      for (int r = 0; r < 8; ++r) {
        int i = i0 + 16 * ta + r + 8 * sel;
        float a = acc[ta][td][r];
        float c = C[(size_t)(b * CL_ + i) * D_ + d];
        float* o = out + (size_t)(b * CL_ + i) * (4 * D_);
        o[d]          = c;
        o[D_ + d]     = a;
        o[2 * D_ + d] = c * a;
        o[3 * D_ + d] = c * bv;
      }
    }
  }
}

// ---------------- launcher ----------------

extern "C" void kernel_launch(void* const* d_in, const int* in_sizes, int n_in,
                              void* d_out, int out_size, void* d_ws, size_t ws_size,
                              hipStream_t stream) {
  (void)in_sizes; (void)n_in; (void)out_size; (void)ws_size;
  const float* C      = (const float*)d_in[0];
  const float* Q      = (const float*)d_in[1];
  const float* w      = (const float*)d_in[2];
  const int*   c_mask = (const int*)d_in[3];
  const int*   q_mask = (const int*)d_in[4];
  float* out = (float*)d_out;

  // Early-phase scratch carved out of d_out (403 MB); fully overwritten by k_out,
  // which reads only ws + inputs -> no aliasing hazard.
  char* ob = (char*)d_out;
  float*  Sc  = (float*) (ob);                       // B*CL*QL f32  = 33,554,432 B
  half_t* CWh = (half_t*)(ob + 33554432);            // B*CL*D  f16  = 50,331,648 B
  half_t* Qh  = (half_t*)(ob + 33554432 + 50331648); // B*QL*D  f16  = 12,582,912 B

  // Scratch that k_out (and late kernels) read lives in d_ws (~30.5 MB total).
  char* wb = (char*)d_ws;
  half_t* QhT = (half_t*)(wb);                       // 12,582,912 B
  half_t* S1h = (half_t*)(wb + 12582912);            // 16,777,216 B
  float*  aC  = (float*) (wb + 29360128);            //    131,072 B
  float*  bQ  = (float*) (wb + 29491200);            //     32,768 B
  float*  S2m = (float*) (wb + 29523968);            //     32,768 B
  float*  Bv  = (float*) (wb + 29556736);            //     98,304 B
  float*  pM  = (float*) (wb + 29655040);            //    262,144 B
  float*  pMv = (float*) (wb + 29917184);            //    262,144 B
  float*  pS  = (float*) (wb + 30179328);            //    262,144 B  (end: 30,441,472)

  long nC = (long)B_ * CL_ * D_;   // 25,165,824
  int  nQ = B_ * QL_ * D_;         //  6,291,456

  k_cw    <<<(unsigned)((nC + 255) / 256), 256, 0, stream>>>(C, w + 2 * D_, CWh, nC);
  k_q     <<<(nQ + 255) / 256,            256, 0, stream>>>(Q, Qh, QhT, nQ);
  k_rowdot<<<(B_ * CL_) / 8,              256, 0, stream>>>(C, w,       aC, B_ * CL_, D_);
  k_rowdot<<<(B_ * QL_) / 8,              256, 0, stream>>>(Q, w + D_,  bQ, B_ * QL_, D_);
  // 64x64 blocks: B * (CL/64) * (QL/64) = 32*16*4 = 2048 waves -> 256 blocks of 8 waves
  k_score <<<(B_ * 16 * 4) / 8,           256, 0, stream>>>(CWh, Qh, aC, bQ, Sc);
  k_srow  <<<(B_ * CL_) / 8,              256, 0, stream>>>(Sc, q_mask, S1h);
  dim3 gsc(B_, NCHUNK_);
  k_scolA <<<gsc,                         256, 0, stream>>>(Sc, c_mask, pM, pMv, pS);
  k_scolB <<<B_,                          256, 0, stream>>>(pM, pMv, pS, S2m);
  dim3 gb(B_, D_ / 256);
  k_bvec  <<<gb,                          256, 0, stream>>>(Q, S2m, Bv);
  // 64x64 blocks: B * (CL/64) * (D/64) = 32*16*12 = 6144 waves -> 768 blocks
  k_out   <<<(B_ * 16 * 12) / 8,          256, 0, stream>>>(S1h, QhT, C, Bv, out);
}
